// AdvancedLogoViT_38740605010379
// MI455X (gfx1250) — matrix operand order repaired and checked by reading, not compiled, not run on hardware
//
#include <hip/hip_runtime.h>
#include <cstdint>
#include <cstddef>

// ---------------------------------------------------------------------------
// Types for CDNA5 WMMA (gfx1250, wave32)
// ---------------------------------------------------------------------------
typedef __attribute__((ext_vector_type(16))) __bf16 v16bf;
typedef __attribute__((ext_vector_type(8)))  float  v8f;
typedef __attribute__((ext_vector_type(4)))  unsigned int u32x4;
typedef __attribute__((ext_vector_type(2)))  unsigned int u32x2;
typedef __attribute__((ext_vector_type(4)))  float  f32x4;

#define WMMA_BF16(a, b, c) \
  __builtin_amdgcn_wmma_f32_16x16x32_bf16(false, (a), false, (b), (short)0, (c), false, false)

__device__ __forceinline__ uint16_t f2bf(float f) {
  uint32_t u = __float_as_uint(f);
  uint32_t r = u + 0x7FFFu + ((u >> 16) & 1u);   // round-to-nearest-even
  return (uint16_t)(r >> 16);
}

#if __has_builtin(__builtin_amdgcn_cvt_pk_bf16_f32)
typedef __attribute__((ext_vector_type(2))) __bf16 v2bf;
__device__ __forceinline__ uint32_t pk2bf(float a, float b) {
  union { v2bf v; uint32_t u; } u;
  u.v = __builtin_amdgcn_cvt_pk_bf16_f32(a, b);
  return u.u;
}
#else
__device__ __forceinline__ uint32_t pk2bf(float a, float b) {
  return (uint32_t)f2bf(a) | ((uint32_t)f2bf(b) << 16);
}
#endif

// CDNA5 async global->LDS copy (ASYNCcnt-tracked), with register-copy fallback.
// Builtin prototype (from hipcc diagnostic): param0 = int4-vector pointer in
// the __device__ (global, AS1) address space; param1 = LDS (AS3) counterpart.
#if __has_builtin(__builtin_amdgcn_global_load_async_to_lds_b128) && \
    __has_builtin(__builtin_amdgcn_s_wait_asynccnt)
#define HAVE_ASYNC_LDS 1
typedef int v4i_vs __attribute__((vector_size(4 * sizeof(int))));
typedef __attribute__((address_space(1))) v4i_vs glb_v4i;
typedef __attribute__((address_space(3))) v4i_vs lds_v4i;
__device__ __forceinline__ void async_cp16(const uint16_t* g, uint16_t* l) {
  __builtin_amdgcn_global_load_async_to_lds_b128((glb_v4i*)g, (lds_v4i*)l, 0, 0);
}
#endif

union FragU { v16bf v; u32x4 q[2]; };

// A fragment: 16(M) x 32(K) bf16. Lane l holds row M=l&15.
// VGPRs 0-3: K = (l<16?0:8)+0..7 ; VGPRs 4-7: K = 16+(l<16?0:8)+0..7
__device__ __forceinline__ v16bf ldfragA(const uint16_t* p0, int rs, int lane) {
  const uint16_t* p = p0 + (lane & 15) * rs + ((lane >> 4) << 3);
  FragU f;
  f.q[0] = *(const u32x4*)p;
  f.q[1] = *(const u32x4*)(p + 16);
  return f.v;
}
// B fragment: 32(K) x 16(N) bf16 with LDS stored as [N][K].
// Lane l holds col N=l&15; VGPRs 0-3: K = (l<16?0:8)+0..7 ; VGPRs 4-7: K = 16+(l<16?0:8)+0..7
__device__ __forceinline__ v16bf ldfragB(const uint16_t* p0, int rs, int lane) {
  const uint16_t* p = p0 + (lane & 15) * rs + ((lane >> 4) << 3);
  FragU f;
  f.q[0] = *(const u32x4*)p;
  f.q[1] = *(const u32x4*)(p + 16);
  return f.v;
}

__device__ __forceinline__ float geluf(float x) {
  return 0.5f * x * (1.0f + erff(x * 0.70710678118654752f));
}
__device__ __forceinline__ float sigmoidf_(float x) { return 1.0f / (1.0f + expf(-x)); }

// ---------------------------------------------------------------------------
// f32 -> bf16 conversion (each element converted exactly once). n4 = count/4
// ---------------------------------------------------------------------------
__global__ void wcvt_kernel(const float* __restrict__ src, uint16_t* __restrict__ dst,
                            long long n4) {
  long long i = (long long)blockIdx.x * 256 + threadIdx.x;
  if (i >= n4) return;
  f32x4 v = *(const f32x4*)(src + i * 4);
  u32x2 p;
  p.x = pk2bf(v.x, v.y);
  p.y = pk2bf(v.z, v.w);
  *(u32x2*)(dst + i * 4) = p;
}

// ---------------------------------------------------------------------------
// Generic GEMM on bf16 operands: C[m][n] = act( A[m][:] . W[n][:] + bias[n] )
// A rows mapped: row -> b=row/a_rpb, t=row%a_rpb, off = a_base + b*a_bs + t*K
// C rows mapped: row -> b=row/c_rpb, t=row%c_rpb, off = c_base + b*c_bs + t*N
// Optional f32 residual (Res, C-mapped); Cf (f32) and/or Cb (bf16) outputs.
// Tile 128x128, BK=32, 256 threads = 8 waves (4x2 wave grid, 32x64 per wave).
// Double-buffered LDS. Out-of-range staging rows/cols are CLAMPED (not
// masked): garbage values only reach C rows/cols that the epilogue masks,
// so staging is branch-free and can use the CDNA5 async global->LDS path.
// ---------------------------------------------------------------------------
__global__ __launch_bounds__(256)
void gemm_bf16_kernel(const uint16_t* __restrict__ A, const uint16_t* __restrict__ W,
                      const float* __restrict__ bias, const float* __restrict__ Res,
                      float* __restrict__ Cf, uint16_t* __restrict__ Cb,
                      int M, int N, int K, int act,
                      int a_rpb, long long a_bs, long long a_base,
                      int c_rpb, long long c_bs, long long c_base) {
  __shared__ uint16_t As[2][128 * 32];
  __shared__ uint16_t Bs[2][128 * 32];

  const int tid  = threadIdx.x;
  const int lane = tid & 31;
  const int w    = tid >> 5;
  const int wm   = w >> 1;       // 0..3 -> M range wm*32 .. +32
  const int wn   = w & 1;        // 0..1 -> N range wn*64 .. +64
  const int brow = blockIdx.x * 128;
  const int bcol = blockIdx.y * 128;

  // Per-thread staging pointers; row mapping (incl. division) done ONCE.
  const uint16_t* aptr[2];
  const uint16_t* bptr[2];
  int sofs[2];
#pragma unroll
  for (int i = 0; i < 2; ++i) {
    int g  = tid + i * 256;        // 0..511
    int r  = g >> 2;               // 0..127
    int kc = (g & 3) << 3;         // 0,8,16,24
    sofs[i] = r * 32 + kc;
    int arow = min(brow + r, M - 1);           // clamp, not mask
    int rb = arow / a_rpb;
    int rt = arow - rb * a_rpb;
    aptr[i] = A + a_base + (long long)rb * a_bs + (long long)rt * K + kc;
    int bcr = min(bcol + r, N - 1);            // clamp, not mask
    bptr[i] = W + (long long)bcr * K + kc;
  }

  v8f acc[2][4] = {};
  const int ksteps = K >> 5;

#if defined(HAVE_ASYNC_LDS)
  // stage kt = 0 via async copies
#pragma unroll
  for (int i = 0; i < 2; ++i) {
    async_cp16(aptr[i], &As[0][sofs[i]]);
    async_cp16(bptr[i], &Bs[0][sofs[i]]);
  }
  __builtin_amdgcn_s_wait_asynccnt(0);
  __syncthreads();

  for (int kt = 0; kt < ksteps; ++kt) {
    const int cur = kt & 1;
    const int nxt = cur ^ 1;
    const bool pre = (kt + 1) < ksteps;

    if (pre) {
      const int off = (kt + 1) * 32;
#pragma unroll
      for (int i = 0; i < 2; ++i) {
        async_cp16(aptr[i] + off, &As[nxt][sofs[i]]);
        async_cp16(bptr[i] + off, &Bs[nxt][sofs[i]]);
      }
    }

    v16bf af[2];
#pragma unroll
    for (int i = 0; i < 2; ++i)
      af[i] = ldfragA(&As[cur][(wm * 32 + i * 16) * 32], 32, lane);
#pragma unroll
    for (int j = 0; j < 4; ++j) {
      v16bf bf = ldfragB(&Bs[cur][(wn * 64 + j * 16) * 32], 32, lane);
#pragma unroll
      for (int i = 0; i < 2; ++i)
        acc[i][j] = WMMA_BF16(af[i], bf, acc[i][j]);
    }

    if (pre) __builtin_amdgcn_s_wait_asynccnt(0);
    __syncthreads();
  }
#else
  // register-staged fallback (branch-free thanks to clamping)
#pragma unroll
  for (int i = 0; i < 2; ++i) {
    *(u32x4*)&As[0][sofs[i]] = *(const u32x4*)(aptr[i]);
    *(u32x4*)&Bs[0][sofs[i]] = *(const u32x4*)(bptr[i]);
  }
  __syncthreads();

  for (int kt = 0; kt < ksteps; ++kt) {
    const int cur = kt & 1;
    const int nxt = cur ^ 1;
    const bool pre = (kt + 1) < ksteps;

    u32x4 va[2], vb[2];
    if (pre) {
      const int off = (kt + 1) * 32;
#pragma unroll
      for (int i = 0; i < 2; ++i) {
        va[i] = *(const u32x4*)(aptr[i] + off);
        vb[i] = *(const u32x4*)(bptr[i] + off);
      }
    }

    v16bf af[2];
#pragma unroll
    for (int i = 0; i < 2; ++i)
      af[i] = ldfragA(&As[cur][(wm * 32 + i * 16) * 32], 32, lane);
#pragma unroll
    for (int j = 0; j < 4; ++j) {
      v16bf bf = ldfragB(&Bs[cur][(wn * 64 + j * 16) * 32], 32, lane);
#pragma unroll
      for (int i = 0; i < 2; ++i)
        acc[i][j] = WMMA_BF16(af[i], bf, acc[i][j]);
    }

    if (pre) {
#pragma unroll
      for (int i = 0; i < 2; ++i) {
        *(u32x4*)&As[nxt][sofs[i]] = va[i];
        *(u32x4*)&Bs[nxt][sofs[i]] = vb[i];
      }
    }
    __syncthreads();
  }
#endif

  // epilogue: bias + activation (+ residual), row-mapped stores
#pragma unroll
  for (int i = 0; i < 2; ++i) {
#pragma unroll
    for (int j = 0; j < 4; ++j) {
      int nl = bcol + wn * 64 + j * 16 + (lane & 15);
      float bv = (bias != nullptr && nl < N) ? bias[nl] : 0.0f;
#pragma unroll
      for (int v2 = 0; v2 < 8; ++v2) {
        int m = brow + wm * 32 + i * 16 + v2 + ((lane >> 4) << 3);
        if (m < M && nl < N) {
          float x = acc[i][j][v2] + bv;
          if (act == 1)      x = fmaxf(x, 0.0f);
          else if (act == 2) x = geluf(x);
          int rb = m / c_rpb;
          int rt = m - rb * c_rpb;
          long long off = c_base + (long long)rb * c_bs + (long long)rt * N + nl;
          if (Res != nullptr) x += Res[off];
          if (Cf != nullptr) Cf[off] = x;
          if (Cb != nullptr) Cb[off] = f2bf(x);
        }
      }
    }
  }
}

// ---------------------------------------------------------------------------
// Fused attention per (head, batch). qkv (bf16) rows: b*NT+t, stride 2304.
// q at col h*64, k at 768+h*64, v at 1536+h*64. Keys padded to 224 (=7*32).
// Output written as bf16 (feeds the projection GEMM).
// ---------------------------------------------------------------------------
#define NKP 224
__global__ __launch_bounds__(256)
void attn_kernel(const uint16_t* __restrict__ qkv, const float* __restrict__ hbias,
                 uint16_t* __restrict__ outp, int NT, int useBias) {
  extern __shared__ char smem[];
  uint16_t* Kt = (uint16_t*)smem;        // [224][64]   keys  (n,d)
  uint16_t* Vt = Kt + NKP * 64;          // [64][224]   V^T   (d,n)
  uint16_t* Qt = Vt + 64 * NKP;          // [32][64]
  uint16_t* Pt = Qt + 32 * 64;           // [32][224]   probs bf16
  float*    Sf = (float*)(Pt + 32 * NKP);// [32][224]   scores f32

  const int tid  = threadIdx.x;
  const int lane = tid & 31;
  const int w    = tid >> 5;
  const int h    = blockIdx.x;
  const int b    = blockIdx.y;

  const uint16_t* qp = qkv + (long long)(b * NT) * 2304 + h * 64;
  const uint16_t* kp = qp + 768;
  const uint16_t* vp = qp + 1536;
  const float biasv = useBias ? hbias[b * 12 + h] : 0.0f;
  const u32x4 zz = {0u, 0u, 0u, 0u};

  // stage K (row-major) and V (transposed), vectorized 8 bf16 at a time
  for (int g = tid; g < NKP * 8; g += 256) {
    int n = g >> 3, db = (g & 7) << 3;
    u32x4 kv = zz, vv = zz;
    if (n < NT) {
      kv = *(const u32x4*)(kp + (long long)n * 2304 + db);
      vv = *(const u32x4*)(vp + (long long)n * 2304 + db);
    }
    *(u32x4*)&Kt[n * 64 + db] = kv;
    const uint16_t* ve = (const uint16_t*)&vv;
#pragma unroll
    for (int j = 0; j < 8; ++j) Vt[(db + j) * NKP + n] = ve[j];
  }
  __syncthreads();

  for (int m0 = 0; m0 < NT; m0 += 32) {
    {
      int m = tid >> 3, db = (tid & 7) << 3;   // 256 groups exactly
      u32x4 qv = zz;
      if (m0 + m < NT) qv = *(const u32x4*)(qp + (long long)(m0 + m) * 2304 + db);
      *(u32x4*)&Qt[m * 64 + db] = qv;
    }
    __syncthreads();

    // S = Q K^T * scale + bias : 32x224 = 2x14 tiles over 8 waves
    for (int t = w; t < 28; t += 8) {
      int mi = t & 1, ni = t >> 1;
      v8f acc = {};
#pragma unroll
      for (int ks = 0; ks < 2; ++ks) {
        v16bf a  = ldfragA(&Qt[(mi * 16) * 64 + ks * 32], 64, lane);
        v16bf bb = ldfragB(&Kt[(ni * 16) * 64 + ks * 32], 64, lane);
        acc = WMMA_BF16(a, bb, acc);
      }
      int n  = ni * 16 + (lane & 15);
      int mb = mi * 16 + ((lane >> 4) << 3);
#pragma unroll
      for (int v2 = 0; v2 < 8; ++v2)
        Sf[(mb + v2) * NKP + n] = acc[v2] * 0.125f + biasv;
    }
    __syncthreads();

    // row softmax: 8 lanes per row, shfl_xor reduce inside wave32
    {
      int r = tid >> 3, sub = tid & 7;
      float vals[28];
      float mx = -1e30f;
#pragma unroll
      for (int c = 0; c < 28; ++c) {
        int n = sub * 28 + c;
        float s = (n < NT) ? Sf[r * NKP + n] : -1e30f;
        vals[c] = s;
        mx = fmaxf(mx, s);
      }
      mx = fmaxf(mx, __shfl_xor(mx, 1));
      mx = fmaxf(mx, __shfl_xor(mx, 2));
      mx = fmaxf(mx, __shfl_xor(mx, 4));
      float sum = 0.f;
#pragma unroll
      for (int c = 0; c < 28; ++c) {
        int n = sub * 28 + c;
        float e = (n < NT) ? expf(vals[c] - mx) : 0.f;
        vals[c] = e;
        sum += e;
      }
      sum += __shfl_xor(sum, 1);
      sum += __shfl_xor(sum, 2);
      sum += __shfl_xor(sum, 4);
      float inv = 1.0f / sum;
#pragma unroll
      for (int c = 0; c < 14; ++c) {
        int n2 = sub * 28 + 2 * c;
        *(uint32_t*)&Pt[r * NKP + n2] = pk2bf(vals[2 * c] * inv, vals[2 * c + 1] * inv);
      }
    }
    __syncthreads();

    // O = P V : 32x64 = 2x4 tiles, one per wave, K=224 in 7 steps
    {
      int mi = w >> 2, ni = w & 3;
      v8f acc = {};
#pragma unroll
      for (int ks = 0; ks < 7; ++ks) {
        v16bf a  = ldfragA(&Pt[(mi * 16) * NKP + ks * 32], NKP, lane);
        v16bf bb = ldfragB(&Vt[(ni * 16) * NKP + ks * 32], NKP, lane);
        acc = WMMA_BF16(a, bb, acc);
      }
      int n  = ni * 16 + (lane & 15);
      int mb = m0 + mi * 16 + ((lane >> 4) << 3);
#pragma unroll
      for (int v2 = 0; v2 < 8; ++v2) {
        int m = mb + v2;
        if (m < NT)
          outp[(long long)(b * NT + m) * 768 + h * 64 + n] = f2bf(acc[v2]);
      }
    }
    __syncthreads();
  }
}

// ---------------------------------------------------------------------------
// LayerNorm over 768, one block per row; writes f32 out + optional bf16 shadow
// ---------------------------------------------------------------------------
__global__ __launch_bounds__(256)
void ln_kernel(const float* __restrict__ in, const float* __restrict__ g,
               const float* __restrict__ be, float* __restrict__ outp,
               uint16_t* __restrict__ outbf, int rows) {
  __shared__ float red[256];
  int r = blockIdx.x;
  if (r >= rows) return;
  int tid = threadIdx.x;
  const float* xp = in + (long long)r * 768;
  float v0 = xp[tid], v1 = xp[tid + 256], v2 = xp[tid + 512];
  red[tid] = v0 + v1 + v2;
  __syncthreads();
  for (int o = 128; o > 0; o >>= 1) { if (tid < o) red[tid] += red[tid + o]; __syncthreads(); }
  float mean = red[0] * (1.0f / 768.0f);
  __syncthreads();
  float d0 = v0 - mean, d1 = v1 - mean, d2 = v2 - mean;
  red[tid] = d0 * d0 + d1 * d1 + d2 * d2;
  __syncthreads();
  for (int o = 128; o > 0; o >>= 1) { if (tid < o) red[tid] += red[tid + o]; __syncthreads(); }
  float rstd = rsqrtf(red[0] * (1.0f / 768.0f) + 1e-5f);
  float o0 = d0 * rstd * g[tid]       + be[tid];
  float o1 = d1 * rstd * g[tid + 256] + be[tid + 256];
  float o2 = d2 * rstd * g[tid + 512] + be[tid + 512];
  float* op = outp + (long long)r * 768;
  op[tid] = o0; op[tid + 256] = o1; op[tid + 512] = o2;
  if (outbf != nullptr) {
    uint16_t* ob = outbf + (long long)r * 768;
    ob[tid] = f2bf(o0); ob[tid + 256] = f2bf(o1); ob[tid + 512] = f2bf(o2);
  }
}

// sf[b][e] = mean over 196 patch tokens of y
__global__ void sf_kernel(const float* __restrict__ y, float* __restrict__ sf) {
  int b = blockIdx.x, chunk = blockIdx.y;
  int e = chunk * 256 + threadIdx.x;
  const float* p = y + ((long long)(b * 197 + 1)) * 768 + e;
  float s = 0.f;
  for (int t = 0; t < 196; ++t) s += p[(long long)t * 768];
  sf[b * 768 + e] = s * (1.0f / 196.0f);
}

// gates + per-head bias for this layer: hbias[b][h]
__global__ __launch_bounds__(256)
void gates_kernel(const float* __restrict__ sf,
                  const float* __restrict__ gw, const float* __restrict__ gb,
                  const float* __restrict__ tw, const float* __restrict__ tb,
                  const float* __restrict__ cw, const float* __restrict__ cb,
                  const float* __restrict__ geo_b, const float* __restrict__ txt_b,
                  const float* __restrict__ col_b, float* __restrict__ hbias) {
  __shared__ float red[3][256];
  __shared__ float gv[3];
  int b = blockIdx.x, tid = threadIdx.x;
  float sg = 0.f, st = 0.f, sc = 0.f;
  for (int e = tid; e < 768; e += 256) {
    float v = sf[b * 768 + e];
    sg += v * gw[e]; st += v * tw[e]; sc += v * cw[e];
  }
  red[0][tid] = sg; red[1][tid] = st; red[2][tid] = sc;
  __syncthreads();
  for (int o = 128; o > 0; o >>= 1) {
    if (tid < o) { red[0][tid] += red[0][tid + o]; red[1][tid] += red[1][tid + o]; red[2][tid] += red[2][tid + o]; }
    __syncthreads();
  }
  if (tid == 0) {
    gv[0] = sigmoidf_(red[0][0] + gb[0]);
    gv[1] = sigmoidf_(red[1][0] + tb[0]);
    gv[2] = sigmoidf_(red[2][0] + cb[0]);
  }
  __syncthreads();
  if (tid < 12)
    hbias[b * 12 + tid] = geo_b[tid] * gv[0] + txt_b[tid] * gv[1] + col_b[tid] * gv[2];
}

// im2col for conv1 (8x8 stride 8 on NCHW 224x224) -> bf16 (25088, 192)
__global__ void im2col1_kernel(const float* __restrict__ x, uint16_t* __restrict__ p1) {
  long long idx = (long long)blockIdx.x * 256 + threadIdx.x;
  if (idx >= 25088LL * 192) return;
  int col = (int)(idx % 192);
  long long row = idx / 192;
  int b = (int)(row / 784);
  int pix = (int)(row % 784);
  int py = pix / 28, px = pix % 28;
  int c = col >> 6, kk = col & 63, ky = kk >> 3, kx = kk & 7;
  p1[idx] = f2bf(x[(((long long)b * 3 + c) * 224 + (py * 8 + ky)) * 224 + (px * 8 + kx)]);
}

// im2col for conv2 (2x2 stride 2 on NHWC 28x28x192 bf16) -> bf16 (6272, 768)
__global__ void im2col2_kernel(const uint16_t* __restrict__ c1o, uint16_t* __restrict__ p2) {
  long long idx = (long long)blockIdx.x * 256 + threadIdx.x;
  if (idx >= 6272LL * 768) return;
  int col = (int)(idx % 768);
  long long row = idx / 768;
  int b = (int)(row / 196);
  int pix = (int)(row % 196);
  int py = pix / 14, px = pix % 14;
  int c = col >> 2, kk = col & 3, ky = kk >> 1, kx = kk & 1;
  p2[idx] = c1o[((long long)b * 784 + (2 * py + ky) * 28 + (2 * px + kx)) * 192 + c];
}

// y[b][0] = cls + pos[0]; y[b][1+t] = patches[b,t] + pos[1+t]; dual f32+bf16
__global__ void assemble_kernel(const float* __restrict__ patches, const float* __restrict__ cls_tok,
                                const float* __restrict__ pos, float* __restrict__ y,
                                uint16_t* __restrict__ ybf) {
  long long idx = (long long)blockIdx.x * 256 + threadIdx.x;
  if (idx >= 6304LL * 768) return;
  int e = (int)(idx % 768);
  long long rt = idx / 768;
  int t = (int)(rt % 197);
  int b = (int)(rt / 197);
  float v = (t == 0) ? cls_tok[e] : patches[((long long)b * 196 + (t - 1)) * 768 + e];
  float o = v + pos[t * 768 + e];
  y[idx] = o;
  ybf[idx] = f2bf(o);
}

// classifier head: out[b][c] = z[b*197] . cls_w[c] + cls_b[c]
__global__ void cls_kernel(const float* __restrict__ z, const float* __restrict__ w,
                           const float* __restrict__ bb, float* __restrict__ outp) {
  int i = threadIdx.x;
  if (i >= 128) return;
  int b = i >> 2, c = i & 3;
  const float* xp = z + (long long)(b * 197) * 768;
  float s = bb[c];
  for (int e = 0; e < 768; ++e) s += xp[e] * w[c * 768 + e];
  outp[i] = s;
}

// ---------------------------------------------------------------------------
// Host orchestration
// ---------------------------------------------------------------------------
static inline void launch_wcvt(hipStream_t s, const float* src, uint16_t* dst, long long n) {
  long long n4 = n >> 2;
  wcvt_kernel<<<(int)((n4 + 255) / 256), 256, 0, s>>>(src, dst, n4);
}

static inline void launch_gemm(hipStream_t s, const uint16_t* A, const uint16_t* W,
                               const float* bias, const float* Res,
                               float* Cf, uint16_t* Cb,
                               int M, int N, int K, int act,
                               int a_rpb, long long a_bs, long long a_base,
                               int c_rpb, long long c_bs, long long c_base) {
  dim3 grid((M + 127) / 128, (N + 127) / 128);
  gemm_bf16_kernel<<<grid, 256, 0, s>>>(A, W, bias, Res, Cf, Cb, M, N, K, act,
                                        a_rpb, a_bs, a_base, c_rpb, c_bs, c_base);
}

extern "C" void kernel_launch(void* const* d_in, const int* in_sizes, int n_in,
                              void* d_out, int out_size, void* d_ws, size_t ws_size,
                              hipStream_t stream) {
  const float* x        = (const float*)d_in[0];
  const float* conv1_w  = (const float*)d_in[1];
  const float* conv1_b  = (const float*)d_in[2];
  const float* conv2_w  = (const float*)d_in[3];
  const float* conv2_b  = (const float*)d_in[4];
  const float* conv3_w  = (const float*)d_in[5];
  const float* conv3_b  = (const float*)d_in[6];
  const float* cls_tok  = (const float*)d_in[7];
  const float* pos      = (const float*)d_in[8];
  const float* mha_w    = (const float*)d_in[9];
  const float* mha_b    = (const float*)d_in[10];
  const float* mha_ow   = (const float*)d_in[11];
  const float* mha_ob   = (const float*)d_in[12];
  const float* ln1_w    = (const float*)d_in[13];
  const float* ln1_b    = (const float*)d_in[14];
  const float* ln2_w    = (const float*)d_in[15];
  const float* ln2_b    = (const float*)d_in[16];
  const float* ff1_w    = (const float*)d_in[17];
  const float* ff1_b    = (const float*)d_in[18];
  const float* ff2_w    = (const float*)d_in[19];
  const float* ff2_b    = (const float*)d_in[20];
  const float* lqkv_w   = (const float*)d_in[21];
  const float* lproj_w  = (const float*)d_in[22];
  const float* lproj_b  = (const float*)d_in[23];
  const float* geo_bias = (const float*)d_in[24];
  const float* txt_bias = (const float*)d_in[25];
  const float* col_bias = (const float*)d_in[26];
  const float* geo_w    = (const float*)d_in[27];
  const float* geo_b    = (const float*)d_in[28];
  const float* txt_w    = (const float*)d_in[29];
  const float* txt_b    = (const float*)d_in[30];
  const float* col_w    = (const float*)d_in[31];
  const float* col_b    = (const float*)d_in[32];
  const float* norm_w   = (const float*)d_in[33];
  const float* norm_b   = (const float*)d_in[34];
  const float* cls_w    = (const float*)d_in[35];
  const float* cls_bv   = (const float*)d_in[36];

  const size_t SEQ = 6304;  // 32*197
  float* ws   = (float*)d_ws;
  float* y    = ws;                          // SEQ*768 f32
  float* z    = y + SEQ * 768;               // SEQ*768 f32 (also conv3 'pat' pre-loop)
  float* sfb  = z + SEQ * 768;               // 32*768 f32
  float* hb   = sfb + (size_t)32 * 768;      // 384 f32
  uint16_t* bfb   = (uint16_t*)(hb + 384);
  uint16_t* ybf   = bfb;                               // SEQ*768
  uint16_t* qkvbf = ybf + SEQ * 768;                   // SEQ*2304
  uint16_t* obf   = qkvbf + SEQ * 2304;                // SEQ*768
  uint16_t* ffhbf = obf + SEQ * 768;                   // SEQ*3072
  uint16_t* wscr  = ffhbf + SEQ * 3072;                // 3072*768 (max weight)
  // conv scratch aliases (used only before the transformer loop)
  uint16_t* p1  = qkvbf;                               // 25088*192
  uint16_t* c1o = qkvbf + (size_t)25088 * 192;         // 25088*192
  uint16_t* p2  = ffhbf;                               // 6272*768
  uint16_t* c2o = ffhbf + (size_t)6272 * 768;          // 6272*384
  float*    pat = z;                                   // 6272*768 f32

  const int ATTN_LDS = (NKP * 64 + 64 * NKP + 32 * 64 + 32 * NKP) * 2 + 32 * NKP * 4; // 104448 B

  // ---- conv stem ----
  im2col1_kernel<<<(int)((25088LL * 192 + 255) / 256), 256, 0, stream>>>(x, p1);
  launch_wcvt(stream, conv1_w, wscr, 192LL * 192);
  launch_gemm(stream, p1, wscr, conv1_b, nullptr, nullptr, c1o, 25088, 192, 192, 2,
              25088, 0, 0, 25088, 0, 0);
  im2col2_kernel<<<(int)((6272LL * 768 + 255) / 256), 256, 0, stream>>>(c1o, p2);
  launch_wcvt(stream, conv2_w, wscr, 384LL * 768);
  launch_gemm(stream, p2, wscr, conv2_b, nullptr, nullptr, c2o, 6272, 384, 768, 2,
              6272, 0, 0, 6272, 0, 0);
  launch_wcvt(stream, conv3_w, wscr, 768LL * 384);
  launch_gemm(stream, c2o, wscr, conv3_b, nullptr, pat, nullptr, 6272, 768, 384, 0,
              6272, 0, 0, 6272, 0, 0);
  assemble_kernel<<<(int)((6304LL * 768 + 255) / 256), 256, 0, stream>>>(pat, cls_tok, pos, y, ybf);

  // ---- transformer layers ----
  for (int l = 0; l < 12; ++l) {
    sf_kernel<<<dim3(32, 3), 256, 0, stream>>>(y, sfb);
    gates_kernel<<<32, 256, 0, stream>>>(sfb, geo_w, geo_b, txt_w, txt_b, col_w, col_b,
                                         geo_bias + l * 12, txt_bias + l * 12,
                                         col_bias + l * 12, hb);
    // global MHA
    launch_wcvt(stream, mha_w + (size_t)l * 2304 * 768, wscr, 2304LL * 768);
    launch_gemm(stream, ybf, wscr, mha_b + (size_t)l * 2304, nullptr, nullptr, qkvbf,
                6304, 2304, 768, 0, 6304, 0, 0, 6304, 0, 0);
    attn_kernel<<<dim3(12, 32), 256, ATTN_LDS, stream>>>(qkvbf, hb, obf, 197, 0);
    launch_wcvt(stream, mha_ow + (size_t)l * 768 * 768, wscr, 768LL * 768);
    launch_gemm(stream, obf, wscr, mha_ob + (size_t)l * 768, y, z, nullptr,
                6304, 768, 768, 0, 6304, 0, 0, 6304, 0, 0);
    ln_kernel<<<6304, 256, 0, stream>>>(z, ln1_w + (size_t)l * 768, ln1_b + (size_t)l * 768,
                                        y, ybf, 6304);
    // FFN
    launch_wcvt(stream, ff1_w + (size_t)l * 3072 * 768, wscr, 3072LL * 768);
    launch_gemm(stream, ybf, wscr, ff1_b + (size_t)l * 3072, nullptr, nullptr, ffhbf,
                6304, 3072, 768, 1, 6304, 0, 0, 6304, 0, 0);
    launch_wcvt(stream, ff2_w + (size_t)l * 768 * 3072, wscr, 768LL * 3072);
    launch_gemm(stream, ffhbf, wscr, ff2_b + (size_t)l * 768, y, z, nullptr,
                6304, 768, 3072, 0, 6304, 0, 0, 6304, 0, 0);
    ln_kernel<<<6304, 256, 0, stream>>>(z, ln2_w + (size_t)l * 768, ln2_b + (size_t)l * 768,
                                        y, ybf, 6304);
    // local (patch-only) attention with gated per-head bias
    launch_wcvt(stream, lqkv_w + (size_t)l * 2304 * 768, wscr, 2304LL * 768);
    launch_gemm(stream, ybf, wscr, nullptr, nullptr, nullptr, qkvbf,
                6272, 2304, 768, 0, 196, 197LL * 768, 768, 6272, 0, 0);
    attn_kernel<<<dim3(12, 32), 256, ATTN_LDS, stream>>>(qkvbf, hb, obf, 196, 1);
    launch_wcvt(stream, lproj_w + (size_t)l * 768 * 768, wscr, 768LL * 768);
    launch_gemm(stream, obf, wscr, lproj_b + (size_t)l * 768, nullptr, y, ybf,
                6272, 768, 768, 0, 6272, 0, 0, 196, 197LL * 768, 768);
  }

  // ---- head ----
  ln_kernel<<<6304, 256, 0, stream>>>(y, norm_w, norm_b, z, nullptr, 6304);
  cls_kernel<<<1, 128, 0, stream>>>(z, cls_w, cls_bv, (float*)d_out);
}
